// GCNModel_75634374083277
// MI455X (gfx1250) — matrix operand order repaired and checked by reading, not compiled, not run on hardware
//
#include <hip/hip_runtime.h>
#include <hip/hip_bf16.h>

// ---------------- problem constants (from reference) ----------------
#define NNODES 50000
#define NEDGES 800000
#define INC    128
#define HID    128
#define OUTC   40
#define OUTP   48   // padded out channels for layer 2 (3 x 16-col tiles)
#define MTILES 3125 // NNODES / 16

// ---------------- workspace layout (bytes, 256-aligned) -------------
#define OFF_DIS  0u                       // N floats                (200,000)
#define OFF_H    200192u                  // N*128 f16               (12,800,000)
#define OFF_XW   13000192u                // N*128 f32               (25,600,000)
#define OFF_ACC  38600192u                // N*128 f32               (25,600,000)
#define OFF_W    64200192u                // 128*128 f16             (32,768)
#define OFF_POOL 64232960u                // 64 f32

typedef __attribute__((ext_vector_type(16))) _Float16 v16h;
typedef __attribute__((ext_vector_type(8)))  _Float16 v8h;
typedef __attribute__((ext_vector_type(8)))  float    v8f;
typedef __attribute__((ext_vector_type(4)))  unsigned int u32x4;
typedef __attribute__((ext_vector_type(8)))  int          i32x8;
typedef __attribute__((ext_vector_type(4)))  int          i32x4;

#if __has_builtin(__builtin_amdgcn_tensor_load_to_lds) && \
    __has_builtin(__builtin_amdgcn_s_wait_tensorcnt)
#define HAVE_TDM 1
#else
#define HAVE_TDM 0
#endif

// ---------------- degree / norm --------------------------------------
__global__ void k_init(float* __restrict__ deg, float* __restrict__ pool, int n) {
  int i = blockIdx.x * blockDim.x + threadIdx.x;
  if (i < n)  deg[i] = 1.0f;            // self-loop
  if (i < 64) pool[i] = 0.0f;
}

__global__ void k_degcount(const int* __restrict__ col, float* __restrict__ deg, int e) {
  int i = blockIdx.x * blockDim.x + threadIdx.x;
  if (i < e) atomicAdd(&deg[col[i]], 1.0f);
}

__global__ void k_rsqrt(float* __restrict__ d, int n) {
  int i = blockIdx.x * blockDim.x + threadIdx.x;
  if (i < n) d[i] = rsqrtf(d[i]);
}

// ---------------- f32 -> f16 conversions -----------------------------
__global__ void k_cvt_h(const float* __restrict__ s, _Float16* __restrict__ d, long long n) {
  long long t = (long long)blockIdx.x * blockDim.x + threadIdx.x;
  if (t < n) d[t] = (_Float16)s[t];
}

// weight [K][C] f32 -> [K][ldB] f16, zero padded past C
__global__ void k_cvt_w(const float* __restrict__ W, _Float16* __restrict__ Wh,
                        int K, int C, int ldB) {
  int t = blockIdx.x * blockDim.x + threadIdx.x;
  if (t >= K * ldB) return;
  int k = t / ldB, c = t % ldB;
  Wh[t] = (c < C) ? (_Float16)W[k * C + c] : (_Float16)0.0f;
}

// ---------------- WMMA GEMM ------------------------------------------
// C[N x LDB] = A[N x 128] * B[128 x LDB].  4 waves / block, each wave owns
// one 16-row tile and sweeps all NT column tiles; B is staged once per block
// into LDS by the Tensor Data Mover (fallback: vectorized copy).
template <int NT, int LDB>
__global__ void __launch_bounds__(128)
k_gemm_wmma(const _Float16* __restrict__ A, const _Float16* __restrict__ B,
            float* __restrict__ C, int mtiles) {
  __shared__ _Float16 Bs[128 * LDB];
  const int tid  = threadIdx.x;
  const int wave = tid >> 5;
  const int lane = tid & 31;

#if HAVE_TDM
  if (wave == 0) {
    // ---- Tensor DMA descriptor: 2-D tensor == tile, 128 x LDB halfs ----
    unsigned long long ga = (unsigned long long)(const void*)B;
    unsigned lds = (unsigned)(unsigned long long)(void*)Bs; // low 32b = LDS offset
    u32x4 g0;
    g0[0] = 1u;                                   // count=1 (valid), user mode
    g0[1] = lds;                                  // lds_addr
    g0[2] = (unsigned)(ga & 0xFFFFFFFFu);         // global_addr[31:0]
    g0[3] = (unsigned)((ga >> 32) & 0x1FFFFFFu)   // global_addr[56:32]
            | (2u << 30);                         // type = 2 ("image")
    i32x8 g1;
    g1[0] = (1 << 16);            // data_size = 1 (2-byte elements)
    g1[1] = (LDB & 0xFFFF) << 16; // tensor_dim0[15:0]
    g1[2] = (128 << 16);          // tensor_dim0[31:16]=0 | tensor_dim1[15:0]=128
    g1[3] = (LDB & 0xFFFF) << 16; // tensor_dim1[31:16]=0 | tile_dim0=LDB
    g1[4] = 128;                  // tile_dim1=128, tile_dim2=0
    g1[5] = LDB;                  // tensor_dim0_stride[31:0]
    g1[6] = 0;                    // stride[47:32] | tensor_dim1_stride[15:0]
    g1[7] = 0;                    // tensor_dim1_stride[47:16]
    i32x4 z4 = {0, 0, 0, 0};
    i32x8 z8 = {0, 0, 0, 0, 0, 0, 0, 0};
    __builtin_amdgcn_tensor_load_to_lds(g0, g1, z4, z4, z8, 0);
    __builtin_amdgcn_s_wait_tensorcnt(0);
  }
  asm volatile("" ::: "memory");   // LDS written behind the compiler's back
#else
  {
    const v8h* gsrc = (const v8h*)B;
    v8h* ldst = (v8h*)Bs;
    for (int i = tid; i < 128 * LDB / 8; i += 128) ldst[i] = gsrc[i];
  }
#endif
  __syncthreads();

  const int tile = blockIdx.x * 4 + wave;
  const int mt   = (tile < mtiles) ? tile : (mtiles - 1);   // clamp for loads
  const int g  = lane >> 4;
  const int mr = lane & 15;

  // ---- A fragments for all 4 K-steps, kept in VGPRs ----
  const _Float16* Arow = A + (size_t)(mt * 16 + mr) * 128;
  v16h a[4];
#pragma unroll
  for (int kt = 0; kt < 4; ++kt) {
    v8h lo = *(const v8h*)(Arow + kt * 32 + g * 8);
    v8h hi = *(const v8h*)(Arow + kt * 32 + 16 + g * 8);
    a[kt] = __builtin_shufflevector(lo, hi,
        0, 1, 2, 3, 4, 5, 6, 7, 8, 9, 10, 11, 12, 13, 14, 15);
  }

#pragma unroll
  for (int nt = 0; nt < NT; ++nt) {
    v8f acc = {};
#pragma unroll
    for (int kt = 0; kt < 4; ++kt) {
      const _Float16* Bcol = Bs + (size_t)(kt * 32) * LDB + nt * 16 + mr;
      v16h b;
#pragma unroll
      for (int s = 0; s < 16; ++s)
        b[s] = Bcol[(size_t)(s < 8 ? 8 * g + s : 16 + 8 * g + (s - 8)) * LDB];
      acc = __builtin_amdgcn_wmma_f32_16x16x32_f16(
          false, a[kt], false, b, (short)0, acc, false, false);
    }
    if (tile < mtiles) {   // wave-uniform, after all WMMAs
#pragma unroll
      for (int r = 0; r < 8; ++r)
        C[(size_t)(mt * 16 + r + 8 * g) * LDB + nt * 16 + mr] = acc[r];
    }
  }
}

// ---------------- self-loop init: acc[i] = dis[i]^2 * xw[i] ----------
__global__ void k_selfloop(const float* __restrict__ dis, const float* __restrict__ xw,
                           float* __restrict__ acc, int n, int c4, int ld) {
  long long t = (long long)blockIdx.x * blockDim.x + threadIdx.x;
  if (t >= (long long)n * c4) return;
  int i = (int)(t / c4);
  int c = (int)(t % c4) * 4;
  float d = dis[i];
  float s = d * d;
  float4 v = *(const float4*)(xw + (size_t)i * ld + c);
  float4 o = make_float4(s * v.x, s * v.y, s * v.z, s * v.w);
  *(float4*)(acc + (size_t)i * ld + c) = o;
}

// ---------------- edge scatter: acc[col] += norm * xw[row] -----------
__global__ void k_edge(const int* __restrict__ row, const int* __restrict__ col,
                       const float* __restrict__ dis, const float* __restrict__ xw,
                       float* __restrict__ acc, int e, int c4, int ld) {
  long long t = (long long)blockIdx.x * blockDim.x + threadIdx.x;
  if (t >= (long long)e * c4) return;
  int ei = (int)(t / c4);
  int c  = (int)(t % c4) * 4;
  int r  = row[ei];
  int cl = col[ei];
  float nrm = dis[r] * dis[cl];
  float4 v = *(const float4*)(xw + (size_t)r * ld + c);
  float* dst = acc + (size_t)cl * ld + c;
  atomicAdd(dst + 0, nrm * v.x);
  atomicAdd(dst + 1, nrm * v.y);
  atomicAdd(dst + 2, nrm * v.z);
  atomicAdd(dst + 3, nrm * v.w);
}

// ---------------- bias + relu + f16 convert (layers 0,1; ld==128) ----
__global__ void k_bias_relu_h(const float* __restrict__ acc, const float* __restrict__ b,
                              _Float16* __restrict__ h, long long total) {
  long long t = (long long)blockIdx.x * blockDim.x + threadIdx.x;
  if (t >= total) return;
  int c = (int)(t & 127);
  float v = acc[t] + b[c];
  h[t] = (_Float16)fmaxf(v, 0.0f);
}

// ---------------- mean pool over nodes (layer 2) ---------------------
__global__ void k_pool(const float* __restrict__ acc, const float* __restrict__ b,
                       float* __restrict__ pool, int n, int ld, int C) {
  int c = threadIdx.x;
  if (c >= C) return;
  float s = 0.0f;
  float bc = b[c];
  for (int i = blockIdx.x; i < n; i += gridDim.x)
    s += acc[(size_t)i * ld + c] + bc;
  atomicAdd(&pool[c], s);
}

__global__ void k_final(const float* __restrict__ pool, float* __restrict__ out,
                        int C, float invN) {
  int c = threadIdx.x;
  if (c < C) out[c] = pool[c] * invN;
}

// ---------------- host driver ---------------------------------------
extern "C" void kernel_launch(void* const* d_in, const int* in_sizes, int n_in,
                              void* d_out, int out_size, void* d_ws, size_t ws_size,
                              hipStream_t stream) {
  (void)in_sizes; (void)n_in; (void)out_size; (void)ws_size;
  const float* x  = (const float*)d_in[0];
  const int*   ei = (const int*)d_in[1];
  const float* W0 = (const float*)d_in[2];
  const float* b0 = (const float*)d_in[3];
  const float* W1 = (const float*)d_in[4];
  const float* b1 = (const float*)d_in[5];
  const float* W2 = (const float*)d_in[6];
  const float* b2 = (const float*)d_in[7];
  float* out = (float*)d_out;

  const int* row = ei;            // edge_index[0] : source
  const int* col = ei + NEDGES;   // edge_index[1] : target

  char* ws = (char*)d_ws;
  float*    dis  = (float*)(ws + OFF_DIS);
  _Float16* hf   = (_Float16*)(ws + OFF_H);
  float*    xw   = (float*)(ws + OFF_XW);
  float*    acc  = (float*)(ws + OFF_ACC);
  _Float16* wh   = (_Float16*)(ws + OFF_W);
  float*    pool = (float*)(ws + OFF_POOL);

  const int B256 = 256;
  const int gemm_blocks = (MTILES + 3) / 4;   // 4 row-tiles (waves) per block

  // ---- normalization coefficients ----
  k_init<<<(NNODES + B256 - 1) / B256, B256, 0, stream>>>(dis, pool, NNODES);
  k_degcount<<<(NEDGES + B256 - 1) / B256, B256, 0, stream>>>(col, dis, NEDGES);
  k_rsqrt<<<(NNODES + B256 - 1) / B256, B256, 0, stream>>>(dis, NNODES);

  // ---- x -> f16 ----
  long long nx = (long long)NNODES * INC;
  k_cvt_h<<<(unsigned)((nx + B256 - 1) / B256), B256, 0, stream>>>(x, hf, nx);

  const float* Ws[3] = {W0, W1, W2};
  const float* bs[3] = {b0, b1, b2};
  for (int layer = 0; layer < 3; ++layer) {
    const int Cout = (layer == 2) ? OUTC : HID;   // real channels
    const int ld   = (layer == 2) ? OUTP : HID;   // padded stride
    const int c4   = Cout / 4;

    // weight -> f16 (padded)
    int wtot = 128 * ld;
    k_cvt_w<<<(wtot + B256 - 1) / B256, B256, 0, stream>>>(Ws[layer], wh, 128, Cout, ld);

    // WMMA GEMM: xw = hf * wh
    if (layer == 2)
      k_gemm_wmma<3, OUTP><<<gemm_blocks, 128, 0, stream>>>(hf, wh, xw, MTILES);
    else
      k_gemm_wmma<8, HID><<<gemm_blocks, 128, 0, stream>>>(hf, wh, xw, MTILES);

    // self-loop init + edge scatter
    long long nsl = (long long)NNODES * c4;
    k_selfloop<<<(unsigned)((nsl + B256 - 1) / B256), B256, 0, stream>>>(
        dis, xw, acc, NNODES, c4, ld);
    long long nes = (long long)NEDGES * c4;
    k_edge<<<(unsigned)((nes + B256 - 1) / B256), B256, 0, stream>>>(
        row, col, dis, xw, acc, NEDGES, c4, ld);

    if (layer < 2) {
      long long tot = (long long)NNODES * HID;
      k_bias_relu_h<<<(unsigned)((tot + B256 - 1) / B256), B256, 0, stream>>>(
          acc, bs[layer], hf, tot);
    } else {
      k_pool<<<512, 64, 0, stream>>>(acc, bs[layer], pool, NNODES, ld, OUTC);
      k_final<<<1, 64, 0, stream>>>(pool, out, OUTC, 1.0f / (float)NNODES);
    }
  }
}
